// AdaptivePredictor_46737834115537
// MI455X (gfx1250) — matrix operand order repaired and checked by reading, not compiled, not run on hardware
//
#include <hip/hip_runtime.h>
#include <math.h>

// ---------------------------------------------------------------------------
// AdaptivePredictor for MI455X (gfx1250, wave32, WMMA bf16 16x16x32)
// B=32 N=2000 D=256 H2=128 H4=64 HORIZON=24  -> 64000 independent rows
// One 256-thread WG (8 waves) owns 128 rows; each wave owns a 16-row M tile
// and runs the whole fused pipeline wave-locally (no barriers in GRU loop).
// ---------------------------------------------------------------------------

typedef __attribute__((ext_vector_type(16))) __bf16 v16bf;
typedef __attribute__((ext_vector_type(8)))  __bf16 v8bf;
typedef __attribute__((ext_vector_type(8)))  float  v8f;

#define LDS_FENCE()   asm volatile("s_wait_dscnt 0" ::: "memory")
#define ASYNC_FENCE() asm volatile("s_wait_asynccnt 0x0" ::: "memory")

// async global->LDS 16-byte copy (CDNA5, ASYNCcnt-tracked, no VGPR round trip)
__device__ __forceinline__ void async_copy16(unsigned lds_byte_addr, const void* gptr) {
  asm volatile("global_load_async_to_lds_b128 %0, %1, off"
               :: "v"(lds_byte_addr), "v"(gptr) : "memory");
}

// workspace layout (bf16 element offsets)
constexpr int OFF_HP  = 0;        // hp_w  128x256
constexpr int OFF_WHH = 32768;    // w_hh  384x128
constexpr int OFF_GO1 = 81920;    // go_w1 64x128
constexpr int OFF_DP1 = 90112;    // dp_w1 256x256
constexpr int OFF_DP2 = 155648;   // dp_w2 padded to 32x256 (rows 24..31 zero)
constexpr int OFF_PG1 = 163840;   // pg_w1 64x256
// end = 180224 bf16 = 360448 bytes

// LDS strides (elements)
constexpr int FS = 264;   // feat / mid stride (256 + 8 pad), bf16
constexpr int HS = 136;   // h / w_hh / go_w1 stride (128 + 8 pad), bf16

// LDS byte offsets
constexpr int L_WHH  = 0;         // 384*136*2 = 104448 (aliased by mid later)
constexpr int L_GO1  = 104448;    // 64*136*2  = 17408
constexpr int L_FEAT = 121856;    // 128*264*2 = 67584
constexpr int L_H    = 189440;    // 128*136*2 = 34816
constexpr int L_GV   = 224256;    // 896 fp32  = 3584  (GRU gate vectors)
constexpr int L_END  = 227840;    // total dynamic LDS bytes

__device__ __forceinline__ float geluf(float x) {
  return 0.5f * x * (1.0f + erff(x * 0.70710678118654752440f));
}
__device__ __forceinline__ float sigmf(float x) {
  return 1.0f / (1.0f + expf(-x));
}

// A fragment (16x32 bf16, M x K) from row-major bf16 buffer with given stride.
__device__ __forceinline__ v16bf frag_a(const __bf16* base, int stride,
                                        int row0, int k0, int lane) {
  int m = lane & 15;
  int q = (lane & 16) ? 8 : 0;
  const __bf16* p = base + (row0 + m) * stride + k0 + q;
  v8bf lo = *(const v8bf*)p;          // K = k0+q .. +7
  v8bf hi = *(const v8bf*)(p + 16);   // K = k0+16+q .. +7
  v16bf a;
#pragma unroll
  for (int i = 0; i < 8; ++i) { a[i] = lo[i]; a[i + 8] = hi[i]; }
  return a;
}

// B fragment (32x16 bf16, K x N) for X @ W^T, W row-major [out][in=stride].
__device__ __forceinline__ v16bf frag_b(const __bf16* w, int stride,
                                        int n0, int k0, int lane) {
  int n  = n0 + (lane & 15);
  int kk = k0 + ((lane & 16) ? 8 : 0);
  const __bf16* p = w + n * stride + kk;
  v8bf lo = *(const v8bf*)p;
  v8bf hi = *(const v8bf*)(p + 16);
  v16bf b;
#pragma unroll
  for (int i = 0; i < 8; ++i) { b[i] = lo[i]; b[i + 8] = hi[i]; }
  return b;
}

__device__ __forceinline__ v8f wmma_bf(v16bf a, v16bf b, v8f c) {
  return __builtin_amdgcn_wmma_f32_16x16x32_bf16(false, a, false, b,
                                                 (short)0, c, false, false);
}

// ---------------------------------------------------------------------------
// prep: fp32 weights -> bf16 in workspace (L2 resident afterwards)
// ---------------------------------------------------------------------------
__global__ void prep_weights(const float* hp_w, const float* w_hh, const float* go_w1,
                             const float* dp_w1, const float* dp_w2, const float* pg_w1,
                             unsigned short* ws_raw) {
  __bf16* ws = (__bf16*)ws_raw;
  int i = blockIdx.x * blockDim.x + threadIdx.x;   // 65536 threads
  if (i < 32768) ws[OFF_HP  + i] = (__bf16)hp_w[i];
  if (i < 49152) ws[OFF_WHH + i] = (__bf16)w_hh[i];
  if (i < 8192)  ws[OFF_GO1 + i] = (__bf16)go_w1[i];
  if (i < 65536) ws[OFF_DP1 + i] = (__bf16)dp_w1[i];
  if (i < 8192)  { int r = i >> 8; ws[OFF_DP2 + i] = (__bf16)((r < 24) ? dp_w2[i] : 0.0f); }
  if (i < 16384) ws[OFF_PG1 + i] = (__bf16)pg_w1[i];
}

// ---------------------------------------------------------------------------
__global__ __launch_bounds__(256)
void predictor_kernel(const float* __restrict__ feat_g, const float* __restrict__ last_g,
                      const float* __restrict__ hp_b,  const float* __restrict__ w_ih,
                      const float* __restrict__ b_ih,  const float* __restrict__ b_hh,
                      const float* __restrict__ go_b1, const float* __restrict__ go_w2,
                      const float* __restrict__ go_b2, const float* __restrict__ dp_b1,
                      const float* __restrict__ dp_b2, const float* __restrict__ pg_b1,
                      const float* __restrict__ pg_w2, const float* __restrict__ pg_b2,
                      const float* __restrict__ logd,
                      const unsigned short* __restrict__ ws_raw,
                      float* __restrict__ out) {
  extern __shared__ char smem[];
  __bf16* whhL  = (__bf16*)(smem + L_WHH);
  __bf16* go1L  = (__bf16*)(smem + L_GO1);
  __bf16* featL = (__bf16*)(smem + L_FEAT);
  __bf16* hL    = (__bf16*)(smem + L_H);
  float*  gv    = (float*)(smem + L_GV);
  __bf16* midL  = (__bf16*)(smem + L_WHH);   // alias: valid after GRU phase

  const __bf16* ws = (const __bf16*)ws_raw;
  const int tid  = threadIdx.x;
  const int lane = tid & 31;
  const int wave = tid >> 5;
  const int g    = (lane >> 4);
  const int c15  = lane & 15;
  const int lRow = wave * 16;
  const long gRow = (long)blockIdx.x * 128 + lRow;

  // ---- async-stage w_hh / go_w1 bf16 -> LDS (overlaps with phase 1) -----
  // 16-byte chunks; rows are 128 elems (16 chunks), padded LDS stride 136.
  for (int c = tid; c < 7168; c += 256) {
    if (c < 6144) {
      int r = c >> 4, co = (c & 15) * 8;
      async_copy16(L_WHH + (unsigned)(r * HS + co) * 2, ws + OFF_WHH + r * 128 + co);
    } else {
      int c2 = c - 6144;
      int r = c2 >> 4, co = (c2 & 15) * 8;
      async_copy16(L_GO1 + (unsigned)(r * HS + co) * 2, ws + OFF_GO1 + r * 128 + co);
    }
  }

  // ---- stage features (wave-local, fp32->bf16) and GRU gate vectors -----
  for (int e = lane; e < 16 * 256; e += 32) {
    int r = e >> 8, c = e & 255;
    featL[(lRow + r) * FS + c] = (__bf16)feat_g[(gRow + r) * 256 + c];
  }
  // gv layout (fp32): [0,384) w_ih | [384,640) b_ih+b_hh (r,z) |
  //                   [640,768) b_ih[256+c] | [768,896) b_hh[256+c]
  for (int e = tid; e < 384; e += 256) gv[e] = w_ih[e];
  if (tid < 256) gv[384 + tid] = b_ih[tid] + b_hh[tid];
  if (tid < 128) { gv[640 + tid] = b_ih[256 + tid]; gv[768 + tid] = b_hh[256 + tid]; }

  float xreg[8];           // GRU scalar input x, per row-group
#pragma unroll
  for (int v = 0; v < 8; ++v) xreg[v] = last_g[gRow + v + 8 * g];

  LDS_FENCE();             // featL ready for this wave

  // ---- phase 1: h0 = feat @ hp_w^T + hp_b (runs under async copies) -----
  v8f hD[8];
#pragma unroll
  for (int jn = 0; jn < 8; ++jn) {
    v8f acc = {};
#pragma unroll
    for (int kt = 0; kt < 8; ++kt) {
      v16bf a = frag_a(featL, FS, lRow, kt * 32, lane);
      v16bf b = frag_b(ws + OFF_HP, 256, jn * 16, kt * 32, lane);
      acc = wmma_bf(a, b, acc);
    }
    float bb = hp_b[jn * 16 + c15];
#pragma unroll
    for (int v = 0; v < 8; ++v) {
      float h0 = acc[v] + bb;
      hD[jn][v] = h0;
      hL[(lRow + v + 8 * g) * HS + jn * 16 + c15] = (__bf16)h0;
    }
  }
  LDS_FENCE();
  ASYNC_FENCE();           // w_hh / go_w1 resident
  __syncthreads();         // ... for all waves

  // ---- phase 2: GRU scan, 24 steps --------------------------------------
  const float gob2 = go_b2[0];
  float gob1r[4], gow2r[4];
#pragma unroll
  for (int jm = 0; jm < 4; ++jm) {
    gob1r[jm] = go_b1[jm * 16 + c15];
    gow2r[jm] = go_w2[jm * 16 + c15];
  }
  float predA[8], predB[8];

  for (int t = 0; t < 24; ++t) {
#pragma unroll
    for (int j = 0; j < 8; ++j) {
      v8f ar = {}, az = {}, an = {};
#pragma unroll
      for (int kt = 0; kt < 4; ++kt) {
        v16bf a  = frag_a(hL, HS, lRow, kt * 32, lane);
        ar = wmma_bf(a, frag_b(whhL, HS,       j * 16, kt * 32, lane), ar);
        az = wmma_bf(a, frag_b(whhL, HS, 128 + j * 16, kt * 32, lane), az);
        an = wmma_bf(a, frag_b(whhL, HS, 256 + j * 16, kt * 32, lane), an);
      }
      int col = j * 16 + c15;
      float wir = gv[col], wiz = gv[128 + col], win = gv[256 + col];
      float brs = gv[384 + col], bzs = gv[512 + col];
      float bin_ = gv[640 + col], bhn = gv[768 + col];
#pragma unroll
      for (int v = 0; v < 8; ++v) {
        float x  = xreg[v];
        float r  = sigmf(x * wir + brs + ar[v]);
        float z  = sigmf(x * wiz + bzs + az[v]);
        float nc = tanhf(x * win + bin_ + r * (an[v] + bhn));
        hD[j][v] = (1.0f - z) * nc + z * hD[j][v];
      }
    }
    // publish h_new (all A-reads of old h are done)
#pragma unroll
    for (int j = 0; j < 8; ++j) {
#pragma unroll
      for (int v = 0; v < 8; ++v)
        hL[(lRow + v + 8 * g) * HS + j * 16 + c15] = (__bf16)hD[j][v];
    }
    LDS_FENCE();
    // pred = gelu(h_new @ go_w1^T + b1) @ go_w2^T + b2
    float part[8] = {0, 0, 0, 0, 0, 0, 0, 0};
    for (int jm = 0; jm < 4; ++jm) {
      v8f am = {};
#pragma unroll
      for (int kt = 0; kt < 4; ++kt) {
        v16bf a = frag_a(hL, HS, lRow, kt * 32, lane);
        am = wmma_bf(a, frag_b(go1L, HS, jm * 16, kt * 32, lane), am);
      }
      float b1 = gob1r[jm], w2 = gow2r[jm];
#pragma unroll
      for (int v = 0; v < 8; ++v) part[v] += geluf(am[v] + b1) * w2;
    }
#pragma unroll
    for (int v = 0; v < 8; ++v) {
      float s = part[v];
      s += __shfl_xor(s, 1); s += __shfl_xor(s, 2);
      s += __shfl_xor(s, 4); s += __shfl_xor(s, 8);
      float pred = s + gob2;                   // known to all 16 lanes of group
      xreg[v] = pred;                          // x_{t+1} = pred_t
      if (t == c15)      predA[v] = pred;      // this lane's output column
      if (t == 16 + c15) predB[v] = pred;
    }
  }

  __syncthreads();   // everyone done with whhL before aliasing it as midL

  // ---- phase 3: direct = gelu(feat @ dp_w1^T + b1) @ dp_w2^T ------------
  for (int jn = 0; jn < 16; ++jn) {
    v8f acc = {};
#pragma unroll
    for (int kt = 0; kt < 8; ++kt) {
      v16bf a = frag_a(featL, FS, lRow, kt * 32, lane);
      acc = wmma_bf(a, frag_b(ws + OFF_DP1, 256, jn * 16, kt * 32, lane), acc);
    }
    float b1 = dp_b1[jn * 16 + c15];
#pragma unroll
    for (int v = 0; v < 8; ++v)
      midL[(lRow + v + 8 * g) * FS + jn * 16 + c15] = (__bf16)geluf(acc[v] + b1);
  }
  LDS_FENCE();
  v8f dir0 = {}, dir1 = {};
#pragma unroll
  for (int kt = 0; kt < 8; ++kt) {
    v16bf a = frag_a(midL, FS, lRow, kt * 32, lane);
    dir0 = wmma_bf(a, frag_b(ws + OFF_DP2, 256, 0,  kt * 32, lane), dir0);
    dir1 = wmma_bf(a, frag_b(ws + OFF_DP2, 256, 16, kt * 32, lane), dir1);
  }

  // ---- phase 4: gate = sigmoid(gelu(feat @ pg_w1^T + b1) @ pg_w2^T + b2)
  float gp[8] = {0, 0, 0, 0, 0, 0, 0, 0};
  for (int jn = 0; jn < 4; ++jn) {
    v8f acc = {};
#pragma unroll
    for (int kt = 0; kt < 8; ++kt) {
      v16bf a = frag_a(featL, FS, lRow, kt * 32, lane);
      acc = wmma_bf(a, frag_b(ws + OFF_PG1, 256, jn * 16, kt * 32, lane), acc);
    }
    float b1 = pg_b1[jn * 16 + c15];
    float w2 = pg_w2[jn * 16 + c15];
#pragma unroll
    for (int v = 0; v < 8; ++v) gp[v] += geluf(acc[v] + b1) * w2;
  }
  const float pgb2 = pg_b2[0];
  float gate8[8];
#pragma unroll
  for (int v = 0; v < 8; ++v) {
    float s = gp[v];
    s += __shfl_xor(s, 1); s += __shfl_xor(s, 2);
    s += __shfl_xor(s, 4); s += __shfl_xor(s, 8);
    gate8[v] = sigmf(s + pgb2);
  }

  // ---- phase 5: blend + decay, write out --------------------------------
  const float dec = expf(logd[0]);
  const float dk0 = expf(-dec * (float)(c15 + 1));
  const float dk1 = expf(-dec * (float)(c15 + 17));
  const float db0 = dp_b2[c15];
  const float db1 = (c15 < 8) ? dp_b2[16 + c15] : 0.0f;
#pragma unroll
  for (int v = 0; v < 8; ++v) {
    long gr = gRow + v + 8 * g;
    float lvv = last_g[gr];
    float gt  = gate8[v];
    {
      float d  = dir0[v] + db0;
      float bl = gt * predA[v] + (1.0f - gt) * d;
      out[gr * 24 + c15] = 0.9f * bl + 0.1f * lvv * dk0;
    }
    if (c15 < 8) {
      float d  = dir1[v] + db1;
      float bl = gt * predB[v] + (1.0f - gt) * d;
      out[gr * 24 + 16 + c15] = 0.9f * bl + 0.1f * lvv * dk1;
    }
  }
}

// ---------------------------------------------------------------------------
extern "C" void kernel_launch(void* const* d_in, const int* in_sizes, int n_in,
                              void* d_out, int out_size, void* d_ws, size_t ws_size,
                              hipStream_t stream) {
  (void)in_sizes; (void)n_in; (void)out_size; (void)ws_size;
  const float* feat  = (const float*)d_in[0];
  const float* last  = (const float*)d_in[1];
  const float* hp_w  = (const float*)d_in[2];
  const float* hp_b  = (const float*)d_in[3];
  const float* w_ih  = (const float*)d_in[4];
  const float* w_hh  = (const float*)d_in[5];
  const float* b_ih  = (const float*)d_in[6];
  const float* b_hh  = (const float*)d_in[7];
  const float* go_w1 = (const float*)d_in[8];
  const float* go_b1 = (const float*)d_in[9];
  const float* go_w2 = (const float*)d_in[10];
  const float* go_b2 = (const float*)d_in[11];
  const float* dp_w1 = (const float*)d_in[12];
  const float* dp_b1 = (const float*)d_in[13];
  const float* dp_w2 = (const float*)d_in[14];
  const float* dp_b2 = (const float*)d_in[15];
  const float* pg_w1 = (const float*)d_in[16];
  const float* pg_b1 = (const float*)d_in[17];
  const float* pg_w2 = (const float*)d_in[18];
  const float* pg_b2 = (const float*)d_in[19];
  const float* logd  = (const float*)d_in[20];
  unsigned short* ws = (unsigned short*)d_ws;
  float* out = (float*)d_out;

  prep_weights<<<256, 256, 0, stream>>>(hp_w, w_hh, go_w1, dp_w1, dp_w2, pg_w1, ws);

  (void)hipFuncSetAttribute((const void*)predictor_kernel,
                            hipFuncAttributeMaxDynamicSharedMemorySize, L_END);
  predictor_kernel<<<500, 256, L_END, stream>>>(
      feat, last, hp_b, w_ih, b_ih, b_hh, go_b1, go_w2, go_b2,
      dp_b1, dp_b2, pg_b1, pg_w2, pg_b2, logd, ws, out);
}
